// MambaBlock_51857435132441
// MI455X (gfx1250) — matrix operand order repaired and checked by reading, not compiled, not run on hardware
//
#include <hip/hip_runtime.h>
#include <hip/hip_bf16.h>
#include <math.h>

// ---------------------------------------------------------------------------
// Bidirectional Mamba block for gfx1250 (MI455X).
// GEMMs: v_wmma_f32_16x16x32_bf16, 32x32 wave tile; weight tiles staged into
//        LDS with GLOBAL_LOAD_ASYNC_TO_LDS_B128 (ASYNCcnt double-buffer).
// Scan:  3-phase chunked linear-recurrence scan (depth 16+64+16, not 1024).
// ---------------------------------------------------------------------------

typedef __attribute__((ext_vector_type(16))) __bf16 v16bf;
typedef __attribute__((ext_vector_type(8)))  __bf16 v8bf;
typedef __attribute__((ext_vector_type(8)))  float  v8f;

#define DEV __device__ __forceinline__

DEV void store_val(float*  p, float v) { *p = v; }
DEV void store_val(__bf16* p, float v) { *p = (__bf16)v; }

DEV float gelu_exact(float v) { return 0.5f * v * (1.0f + erff(v * 0.70710678118654752f)); }
DEV float softplus_f(float v) { return (v > 20.0f) ? v : log1pf(__expf(v)); }
DEV float silu_f(float v)     { return v / (1.0f + __expf(-v)); }

// A fragment (16x32 bf16): lane holds 16 contiguous K values at lh*16.
DEV v16bf load_afrag(const __bf16* p)
{
    return *(const v16bf*)(p);
}
// B fragment: lane holds 16 contiguous K values at lh*16 of row n.
DEV v16bf load_bfrag(const __bf16* p) { return *(const v16bf*)(p); }

DEV v8f wmma_bf16(v16bf a, v16bf b, v8f c)
{
    return __builtin_amdgcn_wmma_f32_16x16x32_bf16(
        false, a, false, b, (short)0, c, false, false);
}

// ---------------------------------------------------------------------------
// Async global -> LDS copy (16 bytes / lane) + ASYNCcnt wait.
// The builtin takes (int4 AS1* gsrc, int4 AS3* lds, imm offset, imm cpol)
// per the compiler's own diagnostic; inline-asm fallback kept for safety.
// ---------------------------------------------------------------------------
#if defined(__has_builtin)
#if __has_builtin(__builtin_amdgcn_global_load_async_to_lds_b128)
#define HAVE_ASYNC_B128 1
#endif
#if __has_builtin(__builtin_amdgcn_s_wait_asynccnt)
#define HAVE_WAIT_ASYNC 1
#endif
#endif

typedef int v4i_vs __attribute__((vector_size(16)));
using as1_v4i = __attribute__((address_space(1))) v4i_vs;
using as3_v4i = __attribute__((address_space(3))) v4i_vs;

DEV void async_b128(__bf16* lds_dst, const __bf16* gsrc)
{
#if HAVE_ASYNC_B128
    __builtin_amdgcn_global_load_async_to_lds_b128(
        (as1_v4i*)(uintptr_t)gsrc,
        (as3_v4i*)(uintptr_t)lds_dst, 0, 0);
#else
    const unsigned la = (unsigned)(uintptr_t)lds_dst;   // low 32 bits = LDS byte addr
    asm volatile("global_load_async_to_lds_b128 %0, %1, off"
                 :: "v"(la), "v"((unsigned long long)(uintptr_t)gsrc) : "memory");
#endif
}

DEV void wait_async0()
{
#if HAVE_WAIT_ASYNC
    __builtin_amdgcn_s_wait_asynccnt(0);
#else
    asm volatile("s_wait_asynccnt 0" ::: "memory");
#endif
}

// ---------------------------------------------------------------------------
// LDS-staged WMMA GEMM: Out[M,N] = act(A[M,K] @ W[N,K]^T + bias[N])
// Wave tile 32x32 (4 accumulators). Block = 8 waves (4M x 2N) = 128x64 tile.
// B tile (64 rows x 64 K) is async-DMA'd to LDS, double-buffered per 64-K
// chunk; the next chunk's DMA overlaps the current chunk's 8 WMMAs.
// A fragments are register double-buffered from global. Requires K % 64 == 0.
// ---------------------------------------------------------------------------
#define BROW 72   // padded LDS row (bf16 elems) to spread banks; 144B, 16B-aligned

template <int ACT, typename OutT>
__global__ __launch_bounds__(256)
void wmma_gemm_lds_kernel(const __bf16* __restrict__ A, const __bf16* __restrict__ W,
                          const float* __restrict__ bias, OutT* __restrict__ Out,
                          int M, int N, int K)
{
    __shared__ alignas(16) __bf16 btile[2][64 * BROW];

    const int lane = threadIdx.x & 31;
    const int wave = threadIdx.x >> 5;
    const int l15  = lane & 15;
    const int lh   = lane >> 4;
    const int tile_m     = blockIdx.y * 128 + (wave & 3) * 32;
    const int tile_n_blk = blockIdx.x * 64;
    const int tile_n     = tile_n_blk + (wave >> 2) * 32;

    const __bf16* ap0 = A + (size_t)(tile_m + l15) * K + lh * 16;
    const __bf16* ap1 = ap0 + (size_t)16 * K;

    // cooperative B staging: thread -> (row sn, 16-elem group sk)
    const int sn = threadIdx.x >> 2;          // 0..63
    const int sk = (threadIdx.x & 3) * 16;    // 0,16,32,48
    const __bf16* wsrc = W + (size_t)(tile_n_blk + sn) * K + sk;
    const int     soff = sn * BROW + sk;

    // this lane's B read rows within the block tile
    const int rb = ((wave >> 2) * 32 + l15) * BROW + lh * 16;

    // stage chunk 0 into buffer 0
    async_b128(&btile[0][0] + soff,     wsrc);
    async_b128(&btile[0][0] + soff + 8, wsrc + 8);
    wait_async0();
    __syncthreads();

    v8f c00 = {}, c01 = {}, c10 = {}, c11 = {};
    v16bf a0 = load_afrag(ap0), a1 = load_afrag(ap1);   // A pipeline, k = 0

    int buf = 0;
    for (int kc = 0; kc < K; kc += 64) {
        const bool more = (kc + 64) < K;
        if (more) {   // overlap next chunk's DMA with this chunk's WMMAs
            __bf16* nd = &btile[buf ^ 1][0] + soff;
            const __bf16* ns = wsrc + kc + 64;
            async_b128(nd,     ns);
            async_b128(nd + 8, ns + 8);
        }
        const __bf16* lb = &btile[buf][0] + rb;
#pragma unroll
        for (int ks = 0; ks < 64; ks += 32) {
            const int ka = kc + ks + 32;
            v16bf a0n = {}, a1n = {};
            if (ka < K) { a0n = load_afrag(ap0 + ka); a1n = load_afrag(ap1 + ka); }
            v16bf b0 = load_bfrag(lb + ks);                // ds_load_b128 x2
            v16bf b1 = load_bfrag(lb + 16 * BROW + ks);
            c00 = wmma_bf16(a0, b0, c00);
            c01 = wmma_bf16(a0, b1, c01);
            c10 = wmma_bf16(a1, b0, c10);
            c11 = wmma_bf16(a1, b1, c11);
            if (ka < K) { a0 = a0n; a1 = a1n; }
        }
        if (more) wait_async0();
        __syncthreads();
        buf ^= 1;
    }

    const int n0 = tile_n + l15;
    const float bv0 = bias ? bias[n0]      : 0.0f;
    const float bv1 = bias ? bias[n0 + 16] : 0.0f;
#pragma unroll
    for (int r = 0; r < 8; ++r) {
        const int m0 = tile_m + r + lh * 8;
        const int m1 = m0 + 16;
        float v00 = c00[r] + bv0, v01 = c01[r] + bv1;
        float v10 = c10[r] + bv0, v11 = c11[r] + bv1;
        if (ACT == 1) { v00 = gelu_exact(v00); v01 = gelu_exact(v01);
                        v10 = gelu_exact(v10); v11 = gelu_exact(v11); }
        else if (ACT == 2) { v00 = softplus_f(v00); v01 = softplus_f(v01);
                             v10 = softplus_f(v10); v11 = softplus_f(v11); }
        store_val(Out + (size_t)m0 * N + n0,      v00);
        store_val(Out + (size_t)m0 * N + n0 + 16, v01);
        store_val(Out + (size_t)m1 * N + n0,      v10);
        store_val(Out + (size_t)m1 * N + n0 + 16, v11);
    }
}

// ---------------------------------------------------------------------------
// Direct-load WMMA GEMM (used for K = 32 dt_proj, where LDS staging is moot).
// ---------------------------------------------------------------------------
template <int ACT, typename OutT>
__global__ __launch_bounds__(256)
void wmma_gemm_kernel(const __bf16* __restrict__ A, const __bf16* __restrict__ W,
                      const float* __restrict__ bias, OutT* __restrict__ Out,
                      int M, int N, int K)
{
    const int lane = threadIdx.x & 31;
    const int wave = threadIdx.x >> 5;
    const int l15  = lane & 15;
    const int lh   = lane >> 4;
    const int tile_m = blockIdx.y * 128 + (wave & 3) * 32;
    const int tile_n = blockIdx.x * 64  + (wave >> 2) * 32;

    const __bf16* ap0 = A + (size_t)(tile_m + l15) * K + lh * 16;
    const __bf16* ap1 = ap0 + (size_t)16 * K;
    const __bf16* wp0 = W + (size_t)(tile_n + l15) * K + lh * 16;
    const __bf16* wp1 = wp0 + (size_t)16 * K;

    v8f c00 = {}, c01 = {}, c10 = {}, c11 = {};
    v16bf a0 = load_afrag(ap0), a1 = load_afrag(ap1);
    v16bf b0 = load_bfrag(wp0), b1 = load_bfrag(wp1);

    int k = 0;
    for (; k + 32 < K; k += 32) {
        const int kn = k + 32;
        if (kn + 32 < K) {
            __builtin_prefetch(ap0 + kn + 32, 0, 1);
            __builtin_prefetch(wp0 + kn + 32, 0, 1);
        }
        v16bf a0n = load_afrag(ap0 + kn);
        v16bf a1n = load_afrag(ap1 + kn);
        v16bf b0n = load_bfrag(wp0 + kn);
        v16bf b1n = load_bfrag(wp1 + kn);
        c00 = wmma_bf16(a0, b0, c00);
        c01 = wmma_bf16(a0, b1, c01);
        c10 = wmma_bf16(a1, b0, c10);
        c11 = wmma_bf16(a1, b1, c11);
        a0 = a0n; a1 = a1n; b0 = b0n; b1 = b1n;
    }
    c00 = wmma_bf16(a0, b0, c00);
    c01 = wmma_bf16(a0, b1, c01);
    c10 = wmma_bf16(a1, b0, c10);
    c11 = wmma_bf16(a1, b1, c11);

    const int n0 = tile_n + l15;
    const float bv0 = bias ? bias[n0]      : 0.0f;
    const float bv1 = bias ? bias[n0 + 16] : 0.0f;
#pragma unroll
    for (int r = 0; r < 8; ++r) {
        const int m0 = tile_m + r + lh * 8;
        const int m1 = m0 + 16;
        float v00 = c00[r] + bv0, v01 = c01[r] + bv1;
        float v10 = c10[r] + bv0, v11 = c11[r] + bv1;
        if (ACT == 1) { v00 = gelu_exact(v00); v01 = gelu_exact(v01);
                        v10 = gelu_exact(v10); v11 = gelu_exact(v11); }
        else if (ACT == 2) { v00 = softplus_f(v00); v01 = softplus_f(v01);
                             v10 = softplus_f(v10); v11 = softplus_f(v11); }
        store_val(Out + (size_t)m0 * N + n0,      v00);
        store_val(Out + (size_t)m0 * N + n0 + 16, v01);
        store_val(Out + (size_t)m1 * N + n0,      v10);
        store_val(Out + (size_t)m1 * N + n0 + 16, v11);
    }
}

// ---------------------------------------------------------------------------
// f32 -> bf16 conversion (weights)
// ---------------------------------------------------------------------------
__global__ __launch_bounds__(256)
void cvt_f32_bf16_kernel(const float* __restrict__ s, __bf16* __restrict__ d, int n)
{
    const int i = blockIdx.x * 256 + threadIdx.x;
    if (i < n) d[i] = (__bf16)s[i];
}

// ---------------------------------------------------------------------------
// Pre-LayerNorm over D=512, writes bf16 in forward order AND time-flipped order.
// ---------------------------------------------------------------------------
__global__ __launch_bounds__(256)
void prenorm_kernel(const float* __restrict__ x, const float* __restrict__ g,
                    const float* __restrict__ beta, __bf16* __restrict__ xnf,
                    __bf16* __restrict__ xnb, int L)
{
    __shared__ float sh[256];
    const int row = blockIdx.x;
    const float* xr = x + (size_t)row * 512;
    const float v0 = xr[threadIdx.x];
    const float v1 = xr[threadIdx.x + 256];

    sh[threadIdx.x] = v0 + v1;
    __syncthreads();
    for (int o = 128; o > 0; o >>= 1) {
        if (threadIdx.x < o) sh[threadIdx.x] += sh[threadIdx.x + o];
        __syncthreads();
    }
    const float mu = sh[0] * (1.0f / 512.0f);
    __syncthreads();
    const float d0 = v0 - mu, d1 = v1 - mu;
    sh[threadIdx.x] = d0 * d0 + d1 * d1;
    __syncthreads();
    for (int o = 128; o > 0; o >>= 1) {
        if (threadIdx.x < o) sh[threadIdx.x] += sh[threadIdx.x + o];
        __syncthreads();
    }
    const float rstd = rsqrtf(sh[0] * (1.0f / 512.0f) + 1e-5f);

    const int b = row / L, t = row % L;
    const size_t rf = (size_t)row * 512;
    const size_t rb = ((size_t)b * L + (L - 1 - t)) * 512;
    int c = threadIdx.x;
    float y0 = d0 * rstd * g[c] + beta[c];
    xnf[rf + c] = (__bf16)y0;  xnb[rb + c] = (__bf16)y0;
    c += 256;
    float y1 = d1 * rstd * g[c] + beta[c];
    xnf[rf + c] = (__bf16)y1;  xnb[rb + c] = (__bf16)y1;
}

// ---------------------------------------------------------------------------
// Causal depthwise conv (K=4) + bias + SiLU on u = xz[:, :1024].
// ---------------------------------------------------------------------------
__global__ __launch_bounds__(256)
void conv_silu_kernel(const float* __restrict__ xz, const float* __restrict__ cw,
                      const float* __restrict__ cb, float* __restrict__ uc,
                      __bf16* __restrict__ ucbf, int L)
{
    const int idx = blockIdx.x * 256 + threadIdx.x;   // over M*1024
    const int d   = idx & 1023;
    const int row = idx >> 10;
    const int t   = row % L;
    float acc = cb[d];
#pragma unroll
    for (int j = 0; j < 4; ++j) {
        const int tt = t - 3 + j;
        if (tt >= 0) acc += cw[d * 4 + j] * xz[(size_t)(row - 3 + j) * 2048 + d];
    }
    const float s = silu_f(acc);
    uc[idx]   = s;
    ucbf[idx] = (__bf16)s;
}

// Extract dt = x_dbl[:, :32] as bf16 for the dt_proj GEMM.
__global__ __launch_bounds__(256)
void slice_dt_kernel(const float* __restrict__ xdbl, __bf16* __restrict__ dt, int M)
{
    const int i = blockIdx.x * 256 + threadIdx.x;     // over M*32
    if (i < M * 32) {
        const int r = i >> 5, c = i & 31;
        dt[i] = (__bf16)xdbl[(size_t)r * 64 + c];
    }
}

// ---------------------------------------------------------------------------
// Chunked selective scan (linear recurrence composition), 3 phases.
// ---------------------------------------------------------------------------
#define T0 16
#define NCHUNK 64

__global__ __launch_bounds__(256)
void scan_phase1_kernel(const float* __restrict__ delta, const float* __restrict__ uc,
                        const float* __restrict__ xdbl, const float* __restrict__ A_log,
                        float* __restrict__ P, float* __restrict__ S, int L)
{
    __shared__ float sB[T0][16];
    const int bc = blockIdx.x >> 2;                   // b*NCHUNK + c
    const int b  = bc >> 6, c = bc & 63;
    const int d  = ((blockIdx.x & 3) << 8) + threadIdx.x;

    {
        const int tl = threadIdx.x >> 4, n = threadIdx.x & 15;
        sB[tl][n] = xdbl[(((size_t)b * L + c * T0 + tl) * 64) + 32 + n];
    }
    __syncthreads();

    float Av[16], Pv[16], Sv[16];
#pragma unroll
    for (int n = 0; n < 16; ++n) {
        Av[n] = -__expf(A_log[(size_t)d * 16 + n]);
        Pv[n] = 1.0f; Sv[n] = 0.0f;
    }
    for (int tl = 0; tl < T0; ++tl) {
        const size_t row = (size_t)b * L + c * T0 + tl;
        const float dl = delta[row * 1024 + d];
        const float du = dl * uc[row * 1024 + d];
#pragma unroll
        for (int n = 0; n < 16; ++n) {
            const float e = __expf(dl * Av[n]);
            Sv[n] = e * Sv[n] + du * sB[tl][n];
            Pv[n] *= e;
        }
    }
    const size_t base = ((size_t)bc * 1024 + d) * 16;
#pragma unroll
    for (int n = 0; n < 16; ++n) { P[base + n] = Pv[n]; S[base + n] = Sv[n]; }
}

__global__ __launch_bounds__(256)
void scan_phase2_kernel(const float* __restrict__ P, const float* __restrict__ S,
                        float* __restrict__ Hprev)
{
    const int tid = blockIdx.x * 256 + threadIdx.x;   // over B*1024*16
    const int n  = tid & 15;
    const int bd = tid >> 4;
    const int b  = bd >> 10, d = bd & 1023;
    float H = 0.0f;
    for (int c = 0; c < NCHUNK; ++c) {
        const size_t idx = (((size_t)(b * NCHUNK + c)) * 1024 + d) * 16 + n;
        Hprev[idx] = H;
        H = P[idx] * H + S[idx];
    }
}

__global__ __launch_bounds__(256)
void scan_phase3_kernel(const float* __restrict__ delta, const float* __restrict__ uc,
                        const float* __restrict__ xdbl, const float* __restrict__ xz,
                        const float* __restrict__ A_log, const float* __restrict__ Dp,
                        const float* __restrict__ Hprev, __bf16* __restrict__ yg, int L)
{
    __shared__ float sB[T0][16], sC[T0][16];
    const int bc = blockIdx.x >> 2;
    const int b  = bc >> 6, c = bc & 63;
    const int d  = ((blockIdx.x & 3) << 8) + threadIdx.x;

    {
        const int tl = threadIdx.x >> 4, n = threadIdx.x & 15;
        const size_t r64 = ((size_t)b * L + c * T0 + tl) * 64;
        sB[tl][n] = xdbl[r64 + 32 + n];
        sC[tl][n] = xdbl[r64 + 48 + n];
    }
    __syncthreads();

    float Av[16], h[16];
    const size_t base = ((size_t)bc * 1024 + d) * 16;
#pragma unroll
    for (int n = 0; n < 16; ++n) {
        Av[n] = -__expf(A_log[(size_t)d * 16 + n]);
        h[n]  = Hprev[base + n];
    }
    const float Dd = Dp[d];

    for (int tl = 0; tl < T0; ++tl) {
        const size_t row = (size_t)b * L + c * T0 + tl;
        const float dl = delta[row * 1024 + d];
        const float u  = uc[row * 1024 + d];
        const float du = dl * u;
        float y = 0.0f;
#pragma unroll
        for (int n = 0; n < 16; ++n) {
            h[n] = __expf(dl * Av[n]) * h[n] + du * sB[tl][n];
            y += h[n] * sC[tl][n];
        }
        y += u * Dd;
        const float z = xz[row * 2048 + 1024 + d];
        yg[row * 1024 + d] = (__bf16)(y * silu_f(z));
    }
}

// ---------------------------------------------------------------------------
// h = LayerNorm_1024( concat(yf, flip(yb)) + concat(x, x) ), written as bf16.
// ---------------------------------------------------------------------------
__global__ __launch_bounds__(256)
void cat_ln_kernel(const float* __restrict__ x, const float* __restrict__ yf,
                   const float* __restrict__ yb, const float* __restrict__ g,
                   const float* __restrict__ beta, __bf16* __restrict__ h, int L)
{
    __shared__ float sh[256];
    const int row = blockIdx.x;
    const int b = row / L, t = row % L;
    const size_t rowb = (size_t)b * L + (L - 1 - t);

    float v[4];
#pragma unroll
    for (int i = 0; i < 4; ++i) {
        const int c = threadIdx.x + i * 256;
        if (c < 512) v[i] = yf[(size_t)row * 512 + c] + x[(size_t)row * 512 + c];
        else         v[i] = yb[rowb * 512 + (c - 512)] + x[(size_t)row * 512 + (c - 512)];
    }
    sh[threadIdx.x] = v[0] + v[1] + v[2] + v[3];
    __syncthreads();
    for (int o = 128; o > 0; o >>= 1) {
        if (threadIdx.x < o) sh[threadIdx.x] += sh[threadIdx.x + o];
        __syncthreads();
    }
    const float mu = sh[0] * (1.0f / 1024.0f);
    __syncthreads();
    float s2 = 0.0f;
#pragma unroll
    for (int i = 0; i < 4; ++i) { const float dd = v[i] - mu; s2 += dd * dd; }
    sh[threadIdx.x] = s2;
    __syncthreads();
    for (int o = 128; o > 0; o >>= 1) {
        if (threadIdx.x < o) sh[threadIdx.x] += sh[threadIdx.x + o];
        __syncthreads();
    }
    const float rstd = rsqrtf(sh[0] * (1.0f / 1024.0f) + 1e-5f);
#pragma unroll
    for (int i = 0; i < 4; ++i) {
        const int c = threadIdx.x + i * 256;
        h[(size_t)row * 1024 + c] = (__bf16)((v[i] - mu) * rstd * g[c] + beta[c]);
    }
}

// ---------------------------------------------------------------------------
// Host-side orchestration
// ---------------------------------------------------------------------------
extern "C" void kernel_launch(void* const* d_in, const int* in_sizes, int n_in,
                              void* d_out, int out_size, void* d_ws, size_t ws_size,
                              hipStream_t stream)
{
    (void)in_sizes; (void)n_in; (void)out_size; (void)ws_size;

    constexpr int B = 2, L = 1024, M = B * L;          // 2048 rows
    constexpr int DM = 512, DI = 1024, DT = 32;

    const float* x = (const float*)d_in[0];
    const float *ipw[2], *cw[2], *cb[2], *xpw[2], *dtw[2], *dtb[2], *alog[2], *Dpar[2], *opw[2];
    for (int dir = 0; dir < 2; ++dir) {
        const int base = 1 + dir * 9;
        ipw[dir]  = (const float*)d_in[base + 0];   // (2048, 512)
        cw[dir]   = (const float*)d_in[base + 1];   // (1024, 4)
        cb[dir]   = (const float*)d_in[base + 2];   // (1024,)
        xpw[dir]  = (const float*)d_in[base + 3];   // (64, 1024)
        dtw[dir]  = (const float*)d_in[base + 4];   // (1024, 32)
        dtb[dir]  = (const float*)d_in[base + 5];   // (1024,)
        alog[dir] = (const float*)d_in[base + 6];   // (1024, 16)
        Dpar[dir] = (const float*)d_in[base + 7];   // (1024,)
        opw[dir]  = (const float*)d_in[base + 8];   // (512, 1024)
    }
    const float* pre_g  = (const float*)d_in[19];
    const float* pre_b  = (const float*)d_in[20];
    const float* ffln_g = (const float*)d_in[21];
    const float* ffln_b = (const float*)d_in[22];
    const float* ff_w1  = (const float*)d_in[23];   // (2048, 1024)
    const float* ff_b1  = (const float*)d_in[24];
    const float* ff_w2  = (const float*)d_in[25];   // (512, 2048)
    const float* ff_b2  = (const float*)d_in[26];

    char* cur = (char*)d_ws;
    auto alloc = [&](size_t bytes) -> void* {
        void* r = (void*)cur;
        cur += (bytes + 255) & ~(size_t)255;
        return r;
    };
    __bf16* xn_bf[2] = { (__bf16*)alloc((size_t)M * DM * 2), (__bf16*)alloc((size_t)M * DM * 2) };
    __bf16 *w_ip[2], *w_xp[2], *w_dt[2], *w_op[2];
    float* yproj[2];
    for (int dir = 0; dir < 2; ++dir) {
        w_ip[dir]  = (__bf16*)alloc((size_t)2 * DI * DM * 2);
        w_xp[dir]  = (__bf16*)alloc((size_t)64 * DI * 2);
        w_dt[dir]  = (__bf16*)alloc((size_t)DI * DT * 2);
        w_op[dir]  = (__bf16*)alloc((size_t)DM * DI * 2);
        yproj[dir] = (float*) alloc((size_t)M * DM * 4);
    }
    float*  xz    = (float*) alloc((size_t)M * 2 * DI * 4);
    float*  ucf   = (float*) alloc((size_t)M * DI * 4);
    __bf16* ucbf  = (__bf16*)alloc((size_t)M * DI * 2);
    float*  xdbl  = (float*) alloc((size_t)M * 64 * 4);
    __bf16* dtbf  = (__bf16*)alloc((size_t)M * DT * 2);
    float*  delta = (float*) alloc((size_t)M * DI * 4);
    __bf16* ygbf  = (__bf16*)alloc((size_t)M * DI * 2);
    float*  Pbuf  = (float*) alloc((size_t)B * NCHUNK * DI * 16 * 4);
    float*  Sbuf  = (float*) alloc((size_t)B * NCHUNK * DI * 16 * 4);
    float*  Hprev = (float*) alloc((size_t)B * NCHUNK * DI * 16 * 4);
    __bf16* w_ff1  = (__bf16*)alloc((size_t)2048 * 1024 * 2);
    __bf16* w_ff2  = (__bf16*)alloc((size_t)512 * 2048 * 2);
    __bf16* h_bf   = (__bf16*)alloc((size_t)M * 1024 * 2);
    __bf16* ff1_bf = (__bf16*)alloc((size_t)M * 2048 * 2);

    auto cvt = [&](const float* s, __bf16* d, int n) {
        cvt_f32_bf16_kernel<<<(n + 255) / 256, 256, 0, stream>>>(s, d, n);
    };

    for (int dir = 0; dir < 2; ++dir) {
        cvt(ipw[dir], w_ip[dir], 2 * DI * DM);
        cvt(xpw[dir], w_xp[dir], 64 * DI);
        cvt(dtw[dir], w_dt[dir], DI * DT);
        cvt(opw[dir], w_op[dir], DM * DI);
    }
    cvt(ff_w1, w_ff1, 2048 * 1024);
    cvt(ff_w2, w_ff2, 512 * 2048);

    prenorm_kernel<<<M, 256, 0, stream>>>(x, pre_g, pre_b, xn_bf[0], xn_bf[1], L);

    for (int dir = 0; dir < 2; ++dir) {
        // in_proj: (M,512) @ (2048,512)^T -> xz (M,2048)
        wmma_gemm_lds_kernel<0, float><<<dim3(2048 / 64, M / 128), 256, 0, stream>>>(
            xn_bf[dir], w_ip[dir], nullptr, xz, M, 2048, 512);
        conv_silu_kernel<<<(M * DI) / 256, 256, 0, stream>>>(
            xz, cw[dir], cb[dir], ucf, ucbf, L);
        // x_proj: (M,1024) @ (64,1024)^T -> x_dbl (M,64)
        wmma_gemm_lds_kernel<0, float><<<dim3(64 / 64, M / 128), 256, 0, stream>>>(
            ucbf, w_xp[dir], nullptr, xdbl, M, 64, 1024);
        slice_dt_kernel<<<(M * DT) / 256, 256, 0, stream>>>(xdbl, dtbf, M);
        // dt_proj + bias + softplus -> delta (M,1024); K=32 -> direct kernel
        wmma_gemm_kernel<2, float><<<dim3(1024 / 64, M / 128), 256, 0, stream>>>(
            dtbf, w_dt[dir], dtb[dir], delta, M, 1024, 32);
        // chunked selective scan + gate
        scan_phase1_kernel<<<B * NCHUNK * (DI / 256), 256, 0, stream>>>(
            delta, ucf, xdbl, alog[dir], Pbuf, Sbuf, L);
        scan_phase2_kernel<<<(B * DI * 16) / 256, 256, 0, stream>>>(Pbuf, Sbuf, Hprev);
        scan_phase3_kernel<<<B * NCHUNK * (DI / 256), 256, 0, stream>>>(
            delta, ucf, xdbl, xz, alog[dir], Dpar[dir], Hprev, ygbf, L);
        // out_proj: (M,1024) @ (512,1024)^T -> yproj (M,512)
        wmma_gemm_lds_kernel<0, float><<<dim3(512 / 64, M / 128), 256, 0, stream>>>(
            ygbf, w_op[dir], nullptr, yproj[dir], M, 512, 1024);
    }

    cat_ln_kernel<<<M, 256, 0, stream>>>(x, yproj[0], yproj[1], ffln_g, ffln_b, h_bf, L);

    wmma_gemm_lds_kernel<1, __bf16><<<dim3(2048 / 64, M / 128), 256, 0, stream>>>(
        h_bf, w_ff1, ff_b1, ff1_bf, M, 2048, 1024);
    wmma_gemm_lds_kernel<0, float><<<dim3(512 / 64, M / 128), 256, 0, stream>>>(
        ff1_bf, w_ff2, ff_b2, (float*)d_out, M, 512, 2048);
}
